// NimbusLinear_65094524338688
// MI455X (gfx1250) — matrix operand order repaired and checked by reading, not compiled, not run on hardware
//
#include <hip/hip_runtime.h>
#include <hip/hip_bf16.h>

typedef __attribute__((ext_vector_type(16))) _Float16 v16h;
typedef __attribute__((ext_vector_type(8)))  _Float16 v8h;
typedef __attribute__((ext_vector_type(8)))  float    v8f;
typedef __attribute__((ext_vector_type(4)))  unsigned long long v4u64;

#define NROWS 32768   // N
#define DFEAT 1024    // D
#define NCODE 64      // C
#define MOUT  1024    // M (output cols)
#define CK    1024    // C*K (GEMM inner dim)

// ---------------------------------------------------------------------------
// Kernel 0: lut f32 -> f16 (one-hot x f16 lut is the exact forward value)
// ---------------------------------------------------------------------------
__global__ void nimbus_lut_cvt(const float* __restrict__ lut,
                               _Float16* __restrict__ lut16, int n) {
    int i = blockIdx.x * blockDim.x + threadIdx.x;
    if (i < n) lut16[i] = (_Float16)lut[i];
}

// ---------------------------------------------------------------------------
// Kernel 1: depth-4 tree traversal -> leaf index (one byte per (n,c))
// ---------------------------------------------------------------------------
__global__ void nimbus_encode(const float* __restrict__ input,
                              const int* __restrict__ dims,
                              const float* __restrict__ thr,
                              unsigned char* __restrict__ leafIdx) {
    int gid = blockIdx.x * blockDim.x + threadIdx.x;   // N*C threads
    int n = gid >> 6;
    int c = gid & 63;
    const float* row = input + (size_t)n * DFEAT;
    int node = 0;
#pragma unroll
    for (int d = 0; d < 4; ++d) {
        int dim = dims[c * 4 + d];
        float x = row[dim];
        float t = thr[c * 15 + node];
        node = 2 * node + 1 + (x > t ? 1 : 0);
    }
    leafIdx[(size_t)n * NCODE + c] = (unsigned char)(node - 15);
}

// ---------------------------------------------------------------------------
// Kernel 2: out(N x M) = OneHot(leafIdx) (N x CK) @ lut16^T (CK x M)
// 128x256 tile per block (8 waves); wave = 4x4 tiles of 16x16; K-step 32.
// A fragments synthesized from leaf indices via 64-bit one-hot shifts.
// B tile staged in LDS column-major so B frags are contiguous ds_load_b128.
// ---------------------------------------------------------------------------
__global__ __launch_bounds__(256)
void nimbus_gemm(const unsigned char* __restrict__ leafIdx,
                 const _Float16* __restrict__ lut16,
                 float* __restrict__ out) {
    __shared__ uint4 sIdxV[512];     //  8 KB: leaf bytes for 128 rows x 64 cbs
    __shared__ v8h   sBv[1024];      // 16 KB: Bt[col][k], 256 cols x 32 halfs

    const int tid      = threadIdx.x;
    const int lane     = tid & 31;
    const int wave     = tid >> 5;
    const int waveRow  = wave >> 2;        // 0..1 -> 64 rows each
    const int waveCol  = wave & 3;         // 0..3 -> 64 cols each
    const int rowBase  = blockIdx.x * 128;
    const int colBase  = blockIdx.y * 256;

    // Stage leaf indices for this row tile: 8192 contiguous bytes.
    {
        const uint4* src = (const uint4*)(leafIdx + (size_t)rowBase * NCODE);
        sIdxV[tid * 2]     = src[tid * 2];
        sIdxV[tid * 2 + 1] = src[tid * 2 + 1];
    }
    const unsigned char* sIdx = (const unsigned char*)sIdxV;

    v8f acc[4][4];
#pragma unroll
    for (int i = 0; i < 4; ++i)
#pragma unroll
        for (int j = 0; j < 4; ++j) acc[i][j] = (v8f)0.0f;

    const unsigned kb0 = (lane & 16) ? 8u : 0u;   // ISA 16-bit A layout

    for (int kb = 0; kb < CK; kb += 32) {
        __syncthreads();   // previous iteration's sB reads complete
        // Stage Bt tile: Bt[col][k] = lut16[(colBase+col)*CK + kb + k]
        {
            const v8h* gsrc = (const v8h*)(lut16 +
                (size_t)(colBase + tid) * CK + kb);
#pragma unroll
            for (int q = 0; q < 4; ++q) sBv[tid * 4 + q] = gsrc[q];
            __builtin_prefetch(gsrc + 4, 0, 0);   // next K-step
        }
        __syncthreads();

        const int c0 = kb >> 4;                  // two codebooks per K-step

        // Build 4 A fragments (one-hot rows): one 64-bit shift + range
        // selects per 4-half group instead of 16 cmp/cndmask pairs.
        v16h afrag[4];
#pragma unroll
        for (int i = 0; i < 4; ++i) {
            int rowLocal = waveRow * 64 + i * 16 + (lane & 15);
            unsigned leaf01 =
                *(const unsigned short*)(sIdx + rowLocal * NCODE + c0);
            unsigned s0 = (leaf01 & 0xffu) - kb0;   // wraps if out of chunk
            unsigned s1 = (leaf01 >> 8)    - kb0;
            unsigned long long oh0 = 0x3c00ULL << ((s0 & 3u) * 16u);
            unsigned long long oh1 = 0x3c00ULL << ((s1 & 3u) * 16u);
            v4u64 t;
            t[0] = (s0 < 4u)      ? oh0 : 0ULL;   // cb0, halfs 0-3
            t[1] = (s0 - 4u < 4u) ? oh0 : 0ULL;   // cb0, halfs 4-7
            t[2] = (s1 < 4u)      ? oh1 : 0ULL;   // cb1, halfs 8-11
            t[3] = (s1 - 4u < 4u) ? oh1 : 0ULL;   // cb1, halfs 12-15
            afrag[i] = __builtin_bit_cast(v16h, t);
        }

        // Load 4 B fragments: lane%16 = column, K = (lane/16)*8 + 0..7, 16 + (lane/16)*8 + 0..7.
        v16h bfrag[4];
#pragma unroll
        for (int j = 0; j < 4; ++j) {
            int colLocal = waveCol * 64 + j * 16 + (lane & 15);
            int khalf    = lane >> 4;
            v8h lo = sBv[colLocal * 4 + khalf];
            v8h hi = sBv[colLocal * 4 + 2 + khalf];
            bfrag[j] = __builtin_shufflevector(lo, hi,
                0, 1, 2, 3, 4, 5, 6, 7, 8, 9, 10, 11, 12, 13, 14, 15);
        }

#pragma unroll
        for (int i = 0; i < 4; ++i)
#pragma unroll
            for (int j = 0; j < 4; ++j)
                acc[i][j] = __builtin_amdgcn_wmma_f32_16x16x32_f16(
                    /*neg_a=*/false, afrag[i],
                    /*neg_b=*/false, bfrag[j],
                    /*c_mod=*/(short)0, acc[i][j],
                    /*reuse_a=*/false, /*reuse_b=*/false);
    }

    // Store: C/D 16x16 f32 layout — VGPR r: M=r (lanes 0-15) / M=8+r (16-31),
    // N = lane%16.
#pragma unroll
    for (int i = 0; i < 4; ++i) {
#pragma unroll
        for (int j = 0; j < 4; ++j) {
            int row0 = rowBase + waveRow * 64 + i * 16 + ((lane & 16) ? 8 : 0);
            int col  = colBase + waveCol * 64 + j * 16 + (lane & 15);
#pragma unroll
            for (int r = 0; r < 8; ++r)
                out[(size_t)(row0 + r) * MOUT + col] = acc[i][j][r];
        }
    }
}

// ---------------------------------------------------------------------------
extern "C" void kernel_launch(void* const* d_in, const int* in_sizes, int n_in,
                              void* d_out, int out_size, void* d_ws, size_t ws_size,
                              hipStream_t stream) {
    const float* input = (const float*)d_in[0];   // (N, D) f32
    const int*   dims  = (const int*)d_in[1];     // (C*4,) i32
    const float* thr   = (const float*)d_in[2];   // (C*15, 1) f32
    const float* lut   = (const float*)d_in[3];   // (M, C, 16) f32
    // d_in[4] selection_matrix, d_in[5] tree_des_mat: structure known, unused.
    float* out = (float*)d_out;                   // (N, M) f32

    _Float16* lut16 = (_Float16*)d_ws;                                  // 2 MB
    unsigned char* leafIdx =
        (unsigned char*)d_ws + (size_t)MOUT * CK * sizeof(_Float16);    // 2 MB

    const int lutN = MOUT * CK;                   // 1,048,576
    nimbus_lut_cvt<<<(lutN + 255) / 256, 256, 0, stream>>>(lut, lut16, lutN);

    const int encN = NROWS * NCODE;               // 2,097,152
    nimbus_encode<<<encN / 256, 256, 0, stream>>>(input, dims, thr, leafIdx);

    dim3 grid(NROWS / 128, MOUT / 256);           // (256, 4)
    nimbus_gemm<<<grid, 256, 0, stream>>>(leafIdx, lut16, out);
}
